// InvariantPointAttention_4183298146713
// MI455X (gfx1250) — matrix operand order repaired and checked by reading, not compiled, not run on hardware
//
#include <hip/hip_runtime.h>
#include <math.h>

#define NRES 512
#define CS   384
#define CZ   128
#define NH   12
#define CHN  16
#define NQP  4
#define NPV  8
#define CATCOL 2112

typedef __attribute__((ext_vector_type(16))) __bf16 v16bf;
typedef __attribute__((ext_vector_type(8)))  float  v8f;
typedef __attribute__((ext_vector_type(2)))  float  v2f;
typedef __attribute__((ext_vector_type(4)))  unsigned int u32x4;
typedef __attribute__((ext_vector_type(8)))  int    i32x8;
typedef __attribute__((ext_vector_type(4)))  int    i32x4;

#if __has_builtin(__builtin_amdgcn_tensor_load_to_lds)
#define HAVE_TDM 1
#else
#define HAVE_TDM 0
#endif

// ---------------- WMMA wrappers ----------------
static __device__ inline v8f wmma_bf16(v16bf a, v16bf b, v8f c) {
  return __builtin_amdgcn_wmma_f32_16x16x32_bf16(false, a, false, b, (short)0, c, false, false);
}
static __device__ inline v8f wmma_f32(v2f a, v2f b, v8f c) {
  return __builtin_amdgcn_wmma_f32_16x16x4_f32(false, a, false, b, (short)0, c, false, false);
}

// ---------------- fragment loaders (ISA 7.12.2 layouts, wave32) ----------------
// NOTE: out-of-range rows/cols are CLAMPED, not zeroed: row m of D depends only on
// row m of A, col n of D only on col n of B, and invalid rows/cols are discarded
// at the store guard. Clamping keeps the inner loops branchless (no saveexec).

// f32 A fragment 16x4: lanes0-15 row M=lane hold K={0,1}; lanes16-31 K={2,3}
static __device__ inline v2f ldA_f32(const float* base, long rs, long cs, int lane, float scale) {
  int m = lane & 15, kh = (lane >> 4) & 1;
  const float* p = base + (long)m * rs + (long)(kh * 2) * cs;
  v2f f; f.x = p[0] * scale; f.y = p[cs] * scale; return f;
}
// f32 B fragment 4x16: lanes0-15 col N=lane hold K={0,1}; lanes16-31 K={2,3}
static __device__ inline v2f ldB_f32(const float* base, long rs, long cs, int lane, int nvalid) {
  int n = lane & 15, kh = (lane >> 4) & 1;
  if (n >= nvalid) n = nvalid - 1;            // clamp (see note above)
  const float* p = base + (long)(kh * 2) * rs + (long)n * cs;
  v2f f; f.x = p[0]; f.y = p[rs]; return f;
}
// bf16 B fragment 32x16 from f32 source: lanes0-15 col N hold K={0..7,16..23}, lanes16-31 K={8..15,24..31}
static __device__ inline v16bf ldB_bf16(const float* base, long rs, long cs, int lane, int nvalid) {
  int n = lane & 15, kh = (lane >> 4) & 1;
  if (n >= nvalid) n = nvalid - 1;            // clamp
  const float* p = base + (long)(kh * 8) * rs + (long)n * cs;
  v16bf f;
#pragma unroll
  for (int i = 0; i < 8; ++i) f[i] = (__bf16)p[i * rs];
#pragma unroll
  for (int i = 0; i < 8; ++i) f[8 + i] = (__bf16)p[(16 + i) * rs];
  return f;
}
// bf16 A fragment 16x32 from bf16 source (contiguous K):
// lanes0-15 row M hold K={0..7,16..23}; lanes16-31 K={8..15,24..31}
static __device__ inline v16bf ldA_bf16src(const __bf16* base, long rs, int lane, int mvalid) {
  int m = lane & 15, kh = (lane >> 4) & 1;
  if (m >= mvalid) m = mvalid - 1;            // clamp
  const __bf16* p = base + (long)m * rs + kh * 8;
  v16bf f;
#pragma unroll
  for (int i = 0; i < 8; ++i) f[i] = p[i];
#pragma unroll
  for (int i = 0; i < 8; ++i) f[8 + i] = p[16 + i];
  return f;
}

// ---------------- generic f32 WMMA GEMM: C = A(MxK) @ B(KxN) [+ bias] ----------------
__global__ __launch_bounds__(128)
void gemm_f32_k(const float* __restrict__ A, const float* __restrict__ B,
                const float* __restrict__ bias, float* __restrict__ C,
                int M, int Nvalid, int K, int lda, int ldb, int ldc, int ntn) {
  int lane = threadIdx.x & 31, wave = threadIdx.x >> 5;
  long tile = (long)blockIdx.x * (blockDim.x >> 5) + wave;
  int mt = (int)(tile / ntn), nt = (int)(tile % ntn);
  if (mt * 16 >= M) return;
  int row0 = mt * 16, col0 = nt * 16;
  v8f acc = {};
  for (int k0 = 0; k0 < K; k0 += 4) {
    v2f a = ldA_f32(A + (long)row0 * lda + k0, lda, 1, lane, 1.f);
    v2f b = ldB_f32(B + (long)k0 * ldb + col0, ldb, 1, lane, Nvalid - col0);
    acc = wmma_f32(a, b, acc);
  }
  int n = lane & 15, rb = ((lane >> 4) & 1) * 8;
  if (col0 + n < Nvalid) {
    float bb = bias ? bias[col0 + n] : 0.f;
#pragma unroll
    for (int r = 0; r < 8; ++r)
      C[(long)(row0 + rb + r) * ldc + col0 + n] = acc[r] + bb;
  }
}

// ---------------- rigid transform of points + squared-norm precompute ----------------
__global__ __launch_bounds__(256)
void point_transform_k(const float* __restrict__ qp, const float* __restrict__ kp,
                       const float* __restrict__ vp, const float* __restrict__ T,
                       float* __restrict__ qpg, float* __restrict__ kpg,
                       float* __restrict__ vpg, float* __restrict__ qs, float* __restrict__ ks) {
  int idx = blockIdx.x * blockDim.x + threadIdx.x;
  if (idx >= NRES * NH) return;
  int i = idx / NH, h = idx % NH;
  const float* Ti = T + i * 16;
  float R00 = Ti[0],  R01 = Ti[1],  R02 = Ti[2],  t0 = Ti[3];
  float R10 = Ti[4],  R11 = Ti[5],  R12 = Ti[6],  t1 = Ti[7];
  float R20 = Ti[8],  R21 = Ti[9],  R22 = Ti[10], t2 = Ti[11];
  float accq = 0.f, acck = 0.f;
#pragma unroll
  for (int p = 0; p < NQP; ++p) {
    long b = (long)i * (NH * NQP * 3) + h * (NQP * 3) + p * 3;
    float lx = qp[b], ly = qp[b + 1], lz = qp[b + 2];
    float gx = R00 * lx + R01 * ly + R02 * lz + t0;
    float gy = R10 * lx + R11 * ly + R12 * lz + t1;
    float gz = R20 * lx + R21 * ly + R22 * lz + t2;
    qpg[b] = gx; qpg[b + 1] = gy; qpg[b + 2] = gz;
    accq += gx * gx + gy * gy + gz * gz;
    lx = kp[b]; ly = kp[b + 1]; lz = kp[b + 2];
    gx = R00 * lx + R01 * ly + R02 * lz + t0;
    gy = R10 * lx + R11 * ly + R12 * lz + t1;
    gz = R20 * lx + R21 * ly + R22 * lz + t2;
    kpg[b] = gx; kpg[b + 1] = gy; kpg[b + 2] = gz;
    acck += gx * gx + gy * gy + gz * gz;
  }
#pragma unroll
  for (int p = 0; p < NPV; ++p) {
    long b = (long)i * (NH * NPV * 3) + h * (NPV * 3) + p * 3;
    float lx = vp[b], ly = vp[b + 1], lz = vp[b + 2];
    vpg[b]     = R00 * lx + R01 * ly + R02 * lz + t0;
    vpg[b + 1] = R10 * lx + R11 * ly + R12 * lz + t1;
    vpg[b + 2] = R20 * lx + R21 * ly + R22 * lz + t2;
  }
  qs[i * NH + h] = accq;
  ks[i * NH + h] = acck;
}

// ---------------- fused attention per (head, 16-row i-tile) ----------------
__global__ __launch_bounds__(128)
void attention_k(const float* __restrict__ q, const float* __restrict__ k,
                 const float* __restrict__ v, const float* __restrict__ qpg,
                 const float* __restrict__ kpg, const float* __restrict__ vpg,
                 const float* __restrict__ qs, const float* __restrict__ ks,
                 const float* __restrict__ biasb, const float* __restrict__ T,
                 const float* __restrict__ gamma, __bf16* __restrict__ a_buf,
                 float* __restrict__ cat) {
  __shared__ float logits[16 * NRES];   // 32 KB: logits, then normalized a (f32)
  __shared__ float red[16 * 8];
  __shared__ float rowstat[16];
  __shared__ float opgS[16 * 32];
  int i0 = blockIdx.x * 16;
  int h = blockIdx.y;
  int tid = threadIdx.x, lane = tid & 31, wave = tid >> 5;

  float g = gamma[h];
  float gsp = (g > 20.f) ? g : log1pf(__expf(g));
  const float wC = 0.23570226039551584f;   // sqrt(2/(9*4))
  const float wL = 0.5773502691896258f;    // sqrt(1/3)
  float sg = gsp * wC;
  float halfs = 0.5f * sg;
  const float invsc = 0.25f;               // 1/sqrt(16)

  // A fragments: q scaled by 1/sqrt(c), qpg scaled by softplus(gamma)*wC
  v2f aq[4], ap[3];
  const float* qbase = q + (long)i0 * (NH * CHN) + h * CHN;
#pragma unroll
  for (int kk = 0; kk < 4; ++kk) aq[kk] = ldA_f32(qbase + kk * 4, NH * CHN, 1, lane, invsc);
  const float* pbase = qpg + (long)i0 * (NH * NQP * 3) + h * (NQP * 3);
#pragma unroll
  for (int kk = 0; kk < 3; ++kk) ap[kk] = ldA_f32(pbase + kk * 4, NH * NQP * 3, 1, lane, sg);

  // phase 1: logits for all j
  for (int jt = wave; jt < NRES / 16; jt += 4) {
    int j0 = jt * 16;
    v8f acc = {};
    const float* kb = k + (long)j0 * (NH * CHN) + h * CHN;
#pragma unroll
    for (int kk = 0; kk < 4; ++kk) {
      v2f b = ldB_f32(kb + kk * 4, 1, NH * CHN, lane, 16);       // k^T
      acc = wmma_f32(aq[kk], b, acc);
    }
    const float* kpb = kpg + (long)j0 * (NH * NQP * 3) + h * (NQP * 3);
#pragma unroll
    for (int kk = 0; kk < 3; ++kk) {
      v2f b = ldB_f32(kpb + kk * 4, 1, NH * NQP * 3, lane, 16);  // kpg^T
      acc = wmma_f32(ap[kk], b, acc);
    }
    int n = lane & 15, rb = ((lane >> 4) & 1) * 8;
    int j = j0 + n;
    float ksj = ks[j * NH + h];
#pragma unroll
    for (int r = 0; r < 8; ++r) {
      int row = rb + r, i = i0 + row;
      float L = wL * (acc[r] + biasb[((long)i * NRES + j) * 16 + h]
                      - halfs * (qs[i * NH + h] + ksj));
      logits[row * NRES + j] = L;
    }
  }
  __syncthreads();

  // phase 2: row softmax (16 rows x 8 segments of 64)
  {
    int r = tid >> 3, sseg = tid & 7;
    float* Lr = logits + r * NRES + sseg * 64;
    float m = -1e30f;
    for (int j = 0; j < 64; ++j) m = fmaxf(m, Lr[j]);
    red[r * 8 + sseg] = m;
    __syncthreads();
    if (sseg == 0) {
      float mm = red[r * 8];
#pragma unroll
      for (int s2 = 1; s2 < 8; ++s2) mm = fmaxf(mm, red[r * 8 + s2]);
      rowstat[r] = mm;
    }
    __syncthreads();
    float mm = rowstat[r], sum = 0.f;
    for (int j = 0; j < 64; ++j) { float e = __expf(Lr[j] - mm); Lr[j] = e; sum += e; }
    red[r * 8 + sseg] = sum;
    __syncthreads();
    if (sseg == 0) {
      float ss = 0.f;
#pragma unroll
      for (int s2 = 0; s2 < 8; ++s2) ss += red[r * 8 + s2];
      rowstat[r] = 1.f / ss;
    }
    __syncthreads();
    float inv = rowstat[r];
    __bf16* ab = a_buf + ((long)h * NRES + i0 + r) * NRES + sseg * 64;
    for (int j = 0; j < 64; ++j) {
      float aa = Lr[j] * inv;
      Lr[j] = aa;                 // normalized a kept in LDS (f32) for phase 3
      ab[j] = (__bf16)aa;         // bf16 copy for o_pair kernel
    }
  }
  __syncthreads();

  // phase 3: o = a@v (wave0), opg = a@vpg (waves 1,2); A from LDS, f32 WMMA K=4
  if (wave < 3) {
    v8f acc = {};
    const float* Bbase = (wave == 0) ? (v + h * CHN)
                                     : (vpg + h * (NPV * 3) + (wave - 1) * 16);
    long ldbv = (wave == 0) ? (long)(NH * CHN) : (long)(NH * NPV * 3);
    int nval = (wave == 2) ? 8 : 16;
    for (int k0 = 0; k0 < NRES; k0 += 4) {
      v2f a = ldA_f32(logits + k0, NRES, 1, lane, 1.f);
      v2f b = ldB_f32(Bbase + (long)k0 * ldbv, ldbv, 1, lane, nval);
      acc = wmma_f32(a, b, acc);
    }
    int n = lane & 15, rb = ((lane >> 4) & 1) * 8;
    if (wave == 0) {
#pragma unroll
      for (int r = 0; r < 8; ++r)
        cat[(long)(i0 + rb + r) * CATCOL + h * CHN + n] = acc[r];
    } else if (n < nval) {
#pragma unroll
      for (int r = 0; r < 8; ++r)
        opgS[(rb + r) * 32 + (wave - 1) * 16 + n] = acc[r];
    }
  }
  __syncthreads();

  // phase 4: op = R^T (opg - t), norms -> cat (16 rows x 8 points)
  {
    int r = tid >> 3, p = tid & 7;
    int i = i0 + r;
    const float* Ti = T + i * 16;
    float ox = opgS[r * 32 + p * 3 + 0] - Ti[3];
    float oy = opgS[r * 32 + p * 3 + 1] - Ti[7];
    float oz = opgS[r * 32 + p * 3 + 2] - Ti[11];
    float o0 = Ti[0] * ox + Ti[4] * oy + Ti[8]  * oz;
    float o1 = Ti[1] * ox + Ti[5] * oy + Ti[9]  * oz;
    float o2 = Ti[2] * ox + Ti[6] * oy + Ti[10] * oz;
    float* cr = cat + (long)i * CATCOL;
    cr[NH * CHN + h * (NPV * 3) + p * 3 + 0] = o0;
    cr[NH * CHN + h * (NPV * 3) + p * 3 + 1] = o1;
    cr[NH * CHN + h * (NPV * 3) + p * 3 + 2] = o2;
    cr[NH * CHN + NH * NPV * 3 + h * NPV + p] = sqrtf(o0 * o0 + o1 * o1 + o2 * o2 + 1e-8f);
  }
}

// ---------------- TDM: stage one 32x128 f32 tile of z[i] into LDS ----------------
#if HAVE_TDM
static __device__ inline void tdm_load_tile(const float* z, int i, int j0, unsigned lds_off) {
  unsigned long long ga =
      (unsigned long long)(const void*)(z + (long)i * NRES * CZ + (long)j0 * CZ);
  u32x4 g0;
  g0[0] = 1u;                                            // count=1, user descriptor
  g0[1] = lds_off;                                       // LDS byte address
  g0[2] = (unsigned)ga;                                  // global_addr lo
  g0[3] = (unsigned)((ga >> 32) & 0x1FFFFFFull) | (2u << 30);  // addr hi | type=2
  i32x8 g1;
  g1[0] = (2 << 16);                                     // data_size = 4B
  g1[1] = (int)((unsigned)CZ << 16);                     // tensor_dim0 = 128 (lo16)
  g1[2] = (int)((unsigned)NRES << 16);                   // dim0 hi | tensor_dim1 = 512
  g1[3] = (int)((unsigned)CZ << 16);                     // dim1 hi | tile_dim0 = 128
  g1[4] = 32;                                            // tile_dim1 = 32, tile_dim2 = 0
  g1[5] = CZ;                                            // tensor_dim0_stride = 128
  g1[6] = 0;
  g1[7] = 0;
  i32x4 gz = {};
#if defined(__clang_major__) && (__clang_major__ >= 23)
  i32x8 gz8 = {};
  __builtin_amdgcn_tensor_load_to_lds(g0, g1, gz, gz, gz8, 0);
#else
  __builtin_amdgcn_tensor_load_to_lds(g0, g1, gz, gz, 0);
#endif
}
#endif

// ---------------- o_pair = a @ z, batched per residue (bf16 WMMA K=32) ----------------
__global__ __launch_bounds__(256)
void opair_k(const __bf16* __restrict__ a_buf, const float* __restrict__ z,
             float* __restrict__ cat) {
  int i = blockIdx.x;
  int lane = threadIdx.x & 31, wave = threadIdx.x >> 5;
  int n0 = wave * 16;                           // 8 waves cover the 128 z channels
  v8f acc = {};
  const __bf16* ab = a_buf + (long)i * NRES;    // rows = heads, stride NRES*NRES
#if HAVE_TDM
  __shared__ float zS[2][32 * CZ];              // double-buffered 16 KB tiles
  const int NT = NRES / 32;
  if (wave == 0) tdm_load_tile(z, i, 0, (unsigned)(size_t)(void*)&zS[0][0]);
  for (int t = 0; t < NT; ++t) {
    if (wave == 0) {
      if (t + 1 < NT) {
        tdm_load_tile(z, i, (t + 1) * 32, (unsigned)(size_t)(void*)&zS[(t + 1) & 1][0]);
        __builtin_amdgcn_s_wait_tensorcnt(1);   // tile t landed
      } else {
        __builtin_amdgcn_s_wait_tensorcnt(0);
      }
    }
    __syncthreads();                            // tile t visible to all waves
    v16bf a = ldA_bf16src(ab + t * 32, (long)NRES * NRES, lane, NH);
    v16bf b = ldB_bf16(&zS[t & 1][0] + n0, CZ, 1, lane, 16);
    acc = wmma_bf16(a, b, acc);
    __syncthreads();                            // protect buffer before overwrite
  }
#else
  const float* zb = z + (long)i * NRES * CZ + n0;
  for (int k0 = 0; k0 < NRES; k0 += 32) {
    if (k0 + 32 < NRES) __builtin_prefetch(zb + (long)(k0 + 32) * CZ, 0, 3);
    v16bf a = ldA_bf16src(ab + k0, (long)NRES * NRES, lane, NH);
    v16bf b = ldB_bf16(zb + (long)k0 * CZ, CZ, 1, lane, 16);
    acc = wmma_bf16(a, b, acc);
  }
#endif
  int n = lane & 15, rb = ((lane >> 4) & 1) * 8;
#pragma unroll
  for (int r = 0; r < 8; ++r) {
    int hh = rb + r;
    if (hh < NH)
      cat[(long)i * CATCOL + (NH * CHN + NH * NPV * 3 + NH * NPV) + hh * CZ + n0 + n] = acc[r];
  }
}

// ---------------- host launcher ----------------
extern "C" void kernel_launch(void* const* d_in, const int* in_sizes, int n_in,
                              void* d_out, int out_size, void* d_ws, size_t ws_size,
                              hipStream_t stream) {
  (void)in_sizes; (void)n_in; (void)out_size; (void)ws_size;
  const float* s     = (const float*)d_in[0];
  const float* z     = (const float*)d_in[1];
  const float* T     = (const float*)d_in[2];
  const float* w_q   = (const float*)d_in[3];
  const float* w_k   = (const float*)d_in[4];
  const float* w_v   = (const float*)d_in[5];
  const float* w_qp  = (const float*)d_in[6];
  const float* w_kp  = (const float*)d_in[7];
  const float* w_vp  = (const float*)d_in[8];
  const float* w_b   = (const float*)d_in[9];
  const float* gamma = (const float*)d_in[10];
  const float* w_out = (const float*)d_in[11];
  const float* b_out = (const float*)d_in[12];
  float* out = (float*)d_out;

  // workspace layout (floats)
  float* f = (float*)d_ws;
  float* q   = f;                       f += (size_t)NRES * NH * CHN;      // 512x192
  float* k   = f;                       f += (size_t)NRES * NH * CHN;
  float* v   = f;                       f += (size_t)NRES * NH * CHN;
  float* qp  = f;                       f += (size_t)NRES * NH * NQP * 3;  // 512x144
  float* kp  = f;                       f += (size_t)NRES * NH * NQP * 3;
  float* vp  = f;                       f += (size_t)NRES * NH * NPV * 3;  // 512x288
  float* qpg = f;                       f += (size_t)NRES * NH * NQP * 3;
  float* kpg = f;                       f += (size_t)NRES * NH * NQP * 3;
  float* vpg = f;                       f += (size_t)NRES * NH * NPV * 3;
  float* qs  = f;                       f += (size_t)NRES * NH;
  float* ks  = f;                       f += (size_t)NRES * NH;
  float* biasb = f;                     f += (size_t)NRES * NRES * 16;     // 16.8 MB
  float* cat   = f;                     f += (size_t)NRES * CATCOL;        // 4.3 MB
  __bf16* a_buf = (__bf16*)f;                                             // 12x512x512 bf16

  auto gemm = [&](const float* A, const float* B, const float* bias, float* C,
                  int M, int Nv, int K, int lda, int ldb, int ldc) {
    int ntn = (Nv + 15) / 16;
    long tiles = (long)(M / 16) * ntn;
    int blocks = (int)((tiles + 3) / 4);
    gemm_f32_k<<<blocks, 128, 0, stream>>>(A, B, bias, C, M, Nv, K, lda, ldb, ldc, ntn);
  };

  // projections
  gemm(s, w_q,  nullptr, q,  NRES, NH * CHN,     CS, CS, NH * CHN,     NH * CHN);
  gemm(s, w_k,  nullptr, k,  NRES, NH * CHN,     CS, CS, NH * CHN,     NH * CHN);
  gemm(s, w_v,  nullptr, v,  NRES, NH * CHN,     CS, CS, NH * CHN,     NH * CHN);
  gemm(s, w_qp, nullptr, qp, NRES, NH * NQP * 3, CS, CS, NH * NQP * 3, NH * NQP * 3);
  gemm(s, w_kp, nullptr, kp, NRES, NH * NQP * 3, CS, CS, NH * NQP * 3, NH * NQP * 3);
  gemm(s, w_vp, nullptr, vp, NRES, NH * NPV * 3, CS, CS, NH * NPV * 3, NH * NPV * 3);

  // rigid transforms + point norms
  point_transform_k<<<(NRES * NH + 255) / 256, 256, 0, stream>>>(
      qp, kp, vp, T, qpg, kpg, vpg, qs, ks);

  // bias = z @ w_b  (rows = (i,j) pairs)
  gemm(z, w_b, nullptr, biasb, NRES * NRES, NH, CZ, CZ, NH, 16);

  // fused attention per (i-tile, head)
  attention_k<<<dim3(NRES / 16, NH), 128, 0, stream>>>(
      q, k, v, qpg, kpg, vpg, qs, ks, biasb, T, gamma, a_buf, cat);

  // o_pair = a @ z per residue (TDM-staged z stream)
  opair_k<<<NRES, 256, 0, stream>>>(a_buf, z, cat);

  // out = cat @ w_out + b_out
  gemm(cat, w_out, b_out, out, NRES, CS, CATCOL, CATCOL, CS, CS);
}
